// MultiHeadAttention_76673756168559
// MI455X (gfx1250) — matrix operand order repaired and checked by reading, not compiled, not run on hardware
//
#include <hip/hip_runtime.h>

// ---- problem constants (from reference) ----
#define HIDDEN 128
#define NHEADS 8
#define DHEAD  16
#define SEQ    2048
#define KB     32      // keys processed per inner iteration
#define WAVES  8       // waves per workgroup (wave32 -> 256 threads)
#define QTILE  16      // query rows per wave
#define SPAD   36      // padded score-row stride (floats): 16B-aligned, low bank conflict
#define F32MIN (-3.40282347e+38f)
#define LOG2E  1.44269504088896340736f

typedef _Float16 v16h __attribute__((ext_vector_type(16)));
typedef _Float16 v8h  __attribute__((ext_vector_type(8)));
typedef float    v8f  __attribute__((ext_vector_type(8)));
typedef float    v4f  __attribute__((ext_vector_type(4)));

// native base-2 exponential: lowers to a single v_exp_f32
__device__ __forceinline__ float fast_exp2(float x) {
    return __builtin_amdgcn_exp2f(x);
}

__global__ __launch_bounds__(WAVES * 32)
void fmha_wmma_kernel(const float* __restrict__ kv,     // [B, S, 2*HIDDEN]
                      const float* __restrict__ q,      // [B, S, HIDDEN]
                      const int*   __restrict__ seq_len,// [B, 1]
                      float* __restrict__ out)          // [B, S, HIDDEN]
{
    // K tile row-major [key][d]  (B-operand of Q*K^T reads contiguous rows)
    __shared__ __align__(16) _Float16 Kt[KB][DHEAD];
    // V tile transposed [d][key] (B-operand of P*V reads contiguous rows)
    __shared__ __align__(16) _Float16 Vt[DHEAD][KB];
    // Per-wave raw-score staging: C-layout -> A-layout transpose
    __shared__ __align__(16) float    Sb[WAVES][QTILE][SPAD];

    const int t    = threadIdx.x;
    const int lane = t & 31;
    const int wave = t >> 5;
    const int qblk = blockIdx.x;
    const int h    = blockIdx.y;
    const int b    = blockIdx.z;

    const int seqlen  = seq_len[b];
    const int nblocks = (seqlen + KB - 1) / KB;   // skip fully-masked blocks (exact)

    const int qbase = qblk * (WAVES * QTILE) + wave * QTILE;
    const int ln16  = lane & 15;
    const int hi    = lane >> 4;                  // 0: lanes 0-15, 1: lanes 16-31

    // ---- A-operand: Q tile 16x32 f16, d padded to K=32 with zeros ----
    // Reference MULTIPLIES Q by sqrt(d_head)=4; fold log2(e) in so softmax
    // runs on exp2 (native v_exp_f32) with exact exp(x-m) semantics.
    v16h aQ;
    {
        const float* qp = q + ((size_t)(b * SEQ + qbase + ln16)) * HIDDEN
                            + h * DHEAD + hi * 8;
        #pragma unroll
        for (int i = 0; i < 8; ++i) aQ[i] = (_Float16)(qp[i] * (4.0f * LOG2E));
        #pragma unroll
        for (int i = 8; i < 16; ++i) aQ[i] = (_Float16)0.0f;
    }

    v8f   acc = {};                 // O accumulator (C-layout: N=d per lane)
    float m = -__builtin_inff();    // per-lane scalar: softmax state of row ln16
    float l = 0.0f;

    for (int kb = 0; kb < nblocks; ++kb) {
        __syncthreads();
        // ---- cooperative K/V fill: 256 threads x (2 K + 2 V floats) ----
        {
            const int key = t >> 3;            // 0..31
            const int d0  = (t & 7) * 2;       // 0,2,...,14
            const float* kp = kv + ((size_t)(b * SEQ + kb * KB + key)) * (2 * HIDDEN)
                                 + h * DHEAD + d0;
            union { _Float16 h2[2]; unsigned u; } pk;
            pk.h2[0] = (_Float16)kp[0];
            pk.h2[1] = (_Float16)kp[1];
            *(unsigned*)&Kt[key][d0] = pk.u;   // one packed 4B LDS store
            const float* vp = kp + HIDDEN;
            Vt[d0][key]     = (_Float16)vp[0]; // transposed scatter
            Vt[d0 + 1][key] = (_Float16)vp[1];
            if (kb + 1 < nblocks)              // warm caches for next block
                __builtin_prefetch(kp + (size_t)KB * 2 * HIDDEN, 0, 1);
        }
        __syncthreads();

        // ---- B-operand: K^T [32(d,padded) x 16 keys]; elems 0-7 = d hi*8.., elems 8-15 = zero pad ----
        v16h bK0 = {}, bK1 = {};
        {
            v8h z  = {};
            v8h a0 = *(const v8h*)&Kt[ln16][hi * 8];
            v8h c0 = *(const v8h*)&Kt[ln16 + 16][hi * 8];
            bK0 = __builtin_shufflevector(a0, z, 0,1,2,3,4,5,6,7,8,9,10,11,12,13,14,15);
            bK1 = __builtin_shufflevector(c0, z, 0,1,2,3,4,5,6,7,8,9,10,11,12,13,14,15);
        }

        // ---- length mask folded into the WMMA C operand (per-lane constant) ----
        const int   col0 = kb * KB + ln16;
        const float add0 = (col0      < seqlen) ? 0.0f : F32MIN;
        const float add1 = (col0 + 16 < seqlen) ? 0.0f : F32MIN;
        v8f c0v, c1v;
        #pragma unroll
        for (int r = 0; r < 8; ++r) { c0v[r] = add0; c1v[r] = add1; }

        v8f s0 = __builtin_amdgcn_wmma_f32_16x16x32_f16(false, aQ, false, bK0,
                                                        (short)0, c0v, false, false);
        v8f s1 = __builtin_amdgcn_wmma_f32_16x16x32_f16(false, aQ, false, bK1,
                                                        (short)0, c1v, false, false);

        // ---- transpose raw scores C-layout -> A-layout through LDS ----
        #pragma unroll
        for (int r = 0; r < 8; ++r) {
            Sb[wave][hi * 8 + r][ln16]      = s0[r];
            Sb[wave][hi * 8 + r][ln16 + 16] = s1[r];
        }
        asm volatile("s_wait_dscnt 0" ::: "memory");   // same-wave store->load

        // lane now reads the 16 values of row ln16 its A-operand slot needs:
        // k in {hi*8..hi*8+7} and {16+hi*8..16+hi*8+7}
        const float* srow = &Sb[wave][ln16][0];
        v4f xa0 = *(const v4f*)(srow + hi * 8);
        v4f xa1 = *(const v4f*)(srow + hi * 8 + 4);
        v4f xb0 = *(const v4f*)(srow + 16 + hi * 8);
        v4f xb1 = *(const v4f*)(srow + 16 + hi * 8 + 4);

        // ---- softmax, fully in-register (one xor-16 shuffle per reduction) ----
        float tm = fmaxf(fmaxf(fmaxf(xa0[0], xa0[1]), fmaxf(xa0[2], xa0[3])),
                         fmaxf(fmaxf(xa1[0], xa1[1]), fmaxf(xa1[2], xa1[3])));
        tm = fmaxf(tm, fmaxf(fmaxf(fmaxf(xb0[0], xb0[1]), fmaxf(xb0[2], xb0[3])),
                             fmaxf(fmaxf(xb1[0], xb1[1]), fmaxf(xb1[2], xb1[3]))));
        tm = fmaxf(tm, __shfl_xor(tm, 16, 32));        // other half of the row
        const float mn = fmaxf(m, tm);
        const float sc = fast_exp2(m - mn);            // first iter: exp2(-inf)=0
        m = mn;

        v16h aP;                                       // A-layout P, built in-register
        float rs = 0.0f;
        #pragma unroll
        for (int i = 0; i < 4; ++i) {
            float p0 = fast_exp2(xa0[i] - mn); rs += p0; aP[i]      = (_Float16)p0;
            float p1 = fast_exp2(xa1[i] - mn); rs += p1; aP[4 + i]  = (_Float16)p1;
            float p2 = fast_exp2(xb0[i] - mn); rs += p2; aP[8 + i]  = (_Float16)p2;
            float p3 = fast_exp2(xb1[i] - mn); rs += p3; aP[12 + i] = (_Float16)p3;
        }
        rs += __shfl_xor(rs, 16, 32);
        l = l * sc + rs;

        // ---- redistribute rescale factor to C-layout rows, rescale O ----
        #pragma unroll
        for (int r = 0; r < 8; ++r) {
            float scr = __shfl(sc, hi * 8 + r, 32);    // row hi*8+r's factor
            acc[r] *= scr;
        }

        // ---- O += P * V (full K=32 contraction, no padding) ----
        v16h bV;
        {
            const int k0 = hi * 8;                     // k hi*8..hi*8+7 and 16+hi*8..16+hi*8+7
            v8h vlo = *(const v8h*)&Vt[ln16][k0];
            v8h vhi = *(const v8h*)&Vt[ln16][k0 + 16];
            bV = __builtin_shufflevector(vlo, vhi, 0,1,2,3,4,5,6,7,8,9,10,11,12,13,14,15);
        }
        acc = __builtin_amdgcn_wmma_f32_16x16x32_f16(false, aP, false, bV,
                                                     (short)0, acc, false, false);
    }

    // ---- epilogue: O = acc / l ; l lives in A-layout, fetch per C-row ----
    #pragma unroll
    for (int r = 0; r < 8; ++r) {
        const float lr  = __shfl(l, hi * 8 + r, 32);
        const int   row = qbase + hi * 8 + r;
        out[((size_t)(b * SEQ + row)) * HIDDEN + h * DHEAD + ln16] = acc[r] / lr;
    }
}

extern "C" void kernel_launch(void* const* d_in, const int* in_sizes, int n_in,
                              void* d_out, int out_size, void* d_ws, size_t ws_size,
                              hipStream_t stream) {
    const float* kv  = (const float*)d_in[0];   // [B, S, 2*HIDDEN] f32
    const float* q   = (const float*)d_in[1];   // [B, S, HIDDEN]   f32
    const int*   sl  = (const int*)d_in[2];     // [B, 1]           i32
    float*       out = (float*)d_out;           // [B, S, HIDDEN]   f32

    const int B = in_sizes[1] / (SEQ * HIDDEN); // = 4
    dim3 grid(SEQ / (WAVES * QTILE), NHEADS, B);
    dim3 block(WAVES * 32);
    fmha_wmma_kernel<<<grid, block, 0, stream>>>(kv, q, sl, out);
}
